// SelfAttention_37890201486014
// MI455X (gfx1250) — matrix operand order repaired and checked by reading, not compiled, not run on hardware
//
#include <hip/hip_runtime.h>

// ---------------------------------------------------------------------------
// CDNA5 (gfx1250, wave32) fused self-attention:
//   K0: one-time fp32 -> bf16 conversion of x and all weight matrices
//   K1: QKV projection  (bf16 in -> Q,K bf16 row-major; V bf16 transposed)
//       4M x 2N register blocking: 8 WMMA per k-step, A tiles reused 2x
//   K2: flash attention (streaming 32 keys/iter, v_wmma_f32_16x16x32_bf16)
//   K3: output projection (same blocking, fp32 result + bias)
// All GEMM inner loops are pure global_load_b128 + v_wmma (no conversions).
// ---------------------------------------------------------------------------

typedef __attribute__((ext_vector_type(16))) __bf16 v16bf;
typedef __attribute__((ext_vector_type(8)))  float  v8f;

union BF16x16 {
    v16bf v;
    uint4 q[2];
    unsigned short h[16];
};

static constexpr int BATCH = 2;
static constexpr int SEQ   = 4096;
static constexpr int HID   = 256;
static constexpr int NHEAD = 8;
static constexpr int HDIM  = 32;

__device__ __forceinline__ unsigned short f2bf(float f) {
    __bf16 b = (__bf16)f;                       // native v_cvt path, RTNE
    return __builtin_bit_cast(unsigned short, b);
}

// ---------------------------------------------------------------------------
// Kernel 0: elementwise fp32 -> bf16 (4 elements / thread, packed stores)
// ---------------------------------------------------------------------------
__global__ __launch_bounds__(256) void cvt_kernel(
    const float* __restrict__ in, unsigned short* __restrict__ out, int n4)
{
    const int i = blockIdx.x * blockDim.x + threadIdx.x;
    if (i >= n4) return;
    const float4 f = *(const float4*)(in + i * 4);
    uint2 p;
    p.x = (unsigned)f2bf(f.x) | ((unsigned)f2bf(f.y) << 16);
    p.y = (unsigned)f2bf(f.z) | ((unsigned)f2bf(f.w) << 16);
    *(uint2*)(out + i * 4) = p;
}

// ---------------------------------------------------------------------------
// Kernel 1: QKV projection.  grid = (M/512, N/32, 3), block = 256 (8 waves).
// Each wave computes a 64x32 strip: 4 M-tiles x 2 N-tiles; every A tile is
// reused by both B columns (8 WMMA per 12 b128 loads per k-step).
// mat 0 -> Q (bf16 row-major), 1 -> K (bf16 row-major), 2 -> V transposed.
// ---------------------------------------------------------------------------
__global__ __launch_bounds__(256) void qkv_kernel(
    const unsigned short* __restrict__ xbf,
    const unsigned short* __restrict__ wqbf, const float* __restrict__ bq,
    const unsigned short* __restrict__ wkbf, const float* __restrict__ bk,
    const unsigned short* __restrict__ wvbf, const float* __restrict__ bv,
    unsigned short* __restrict__ Qbf,
    unsigned short* __restrict__ Kbf,
    unsigned short* __restrict__ Vt)
{
    const int lane  = threadIdx.x & 31;
    const int wave  = threadIdx.x >> 5;
    const int Mbase = blockIdx.x * 512 + wave * 64;
    const int Nbase = blockIdx.y * 32;
    const int mat   = blockIdx.z;

    const unsigned short* W = (mat == 0) ? wqbf : (mat == 1) ? wkbf : wvbf;
    const float* bias       = (mat == 0) ? bq   : (mat == 1) ? bk   : bv;

    const int lhi = lane >> 4;
    const int llo = lane & 15;

    v8f acc[4][2] = {};

#pragma unroll
    for (int k0 = 0; k0 < HID; k0 += 32) {
        // ---- 2 B tiles (cols Nbase..+15 and +16..+31), shared by 4 M-tiles ----
        BF16x16 B0, B1;
        {
            const unsigned short* bp0 = W + (Nbase + llo)      * HID + k0 + (lhi << 4);
            const unsigned short* bp1 = W + (Nbase + 16 + llo) * HID + k0 + (lhi << 4);
            B0.q[0] = *(const uint4*)(bp0);  B0.q[1] = *(const uint4*)(bp0 + 8);
            B1.q[0] = *(const uint4*)(bp1);  B1.q[1] = *(const uint4*)(bp1 + 8);
        }

        // ---- 4 A tiles, contiguous k layout (k = 16h + s) ----
        const int kb = k0 + (lhi << 4);
        BF16x16 A[4];
        {
            const unsigned short* ap = xbf + (Mbase + llo) * HID + kb;
#pragma unroll
            for (int t = 0; t < 4; ++t) {
                A[t].q[0] = *(const uint4*)(ap);
                A[t].q[1] = *(const uint4*)(ap + 8);
                ap += 16 * HID;
            }
        }

#pragma unroll
        for (int t = 0; t < 4; ++t) {
            acc[t][0] = __builtin_amdgcn_wmma_f32_16x16x32_bf16(
                false, A[t].v, false, B0.v, (short)0, acc[t][0], false, false);
            acc[t][1] = __builtin_amdgcn_wmma_f32_16x16x32_bf16(
                false, A[t].v, false, B1.v, (short)0, acc[t][1], false, false);
        }
    }

#pragma unroll
    for (int u = 0; u < 2; ++u) {
        const int bcol = Nbase + u * 16 + llo;
        const float bn = bias[bcol];
#pragma unroll
        for (int t = 0; t < 4; ++t) {
#pragma unroll
            for (int r = 0; r < 8; ++r) {
                const float val = acc[t][u][r] + bn;
                const int m = Mbase + t * 16 + r + (lhi << 3);
                if (mat < 2) {
                    unsigned short* out = (mat == 0) ? Qbf : Kbf;
                    out[m * HID + bcol] = f2bf(val);
                } else {
                    const int b    = m >> 12;
                    const int srow = m & (SEQ - 1);
                    const int hh   = bcol >> 5;
                    const int d    = bcol & 31;
                    Vt[((b * NHEAD + hh) * HDIM + d) * SEQ + srow] = f2bf(val);
                }
            }
        }
    }
}

// ---------------------------------------------------------------------------
// Kernel 2: flash attention.  grid = (SEQ/128, BATCH*NHEAD), block = 256.
// Each wave owns 16 query rows; streams 32 keys/iter:
//   2 score WMMAs -> online softmax -> LDS bounce -> 2 PV WMMAs.
// Output written as bf16 row-major (feeds the projection GEMM directly).
// ---------------------------------------------------------------------------
__global__ __launch_bounds__(256) void attn_kernel(
    const unsigned short* __restrict__ Qbf,
    const unsigned short* __restrict__ Kbf,
    const unsigned short* __restrict__ Vt,
    const unsigned char*  __restrict__ mask,
    unsigned short* __restrict__ Obf)
{
    __shared__ float plds[8][16][36];

    const int lane = threadIdx.x & 31;
    const int wave = threadIdx.x >> 5;
    const int lhi  = lane >> 4;
    const int llo  = lane & 15;

    const int bh = blockIdx.y;
    const int b  = bh >> 3;
    const int h  = bh & 7;
    const int qb = (blockIdx.x * 8 + wave) * 16;

    BF16x16 Q;
    {
        const int grow = (b * SEQ + qb + llo) * HID + h * HDIM;
        const int kb   = lhi << 4;
        Q.q[0] = *(const uint4*)(Qbf + grow + kb);
        Q.q[1] = *(const uint4*)(Qbf + grow + kb + 8);
    }

    v8f o0 = {}, o1 = {};
    float mrow[8], lrow[8];
#pragma unroll
    for (int r = 0; r < 8; ++r) { mrow[r] = -3.0e38f; lrow[r] = 0.0f; }

    const float scale = 0.17677669529663687f;   // 1/sqrt(32)
    const unsigned short* Vbase = Vt + (b * NHEAD + h) * HDIM * SEQ;

    for (int jb = 0; jb < SEQ; jb += 32) {
        const int j0   = jb + llo;
        const int dofs = lhi << 4;
        const unsigned short* kp0 = Kbf + (b * SEQ + j0) * HID + h * HDIM + dofs;
        const unsigned short* kp1 = kp0 + 16 * HID;
        BF16x16 K0, K1;
        K0.q[0] = *(const uint4*)(kp0);  K0.q[1] = *(const uint4*)(kp0 + 8);
        K1.q[0] = *(const uint4*)(kp1);  K1.q[1] = *(const uint4*)(kp1 + 8);

        const int jofs = lhi << 4;
        const unsigned short* vp0 = Vbase + (llo)      * SEQ + jb + jofs;
        const unsigned short* vp1 = Vbase + (llo + 16) * SEQ + jb + jofs;
        BF16x16 V0, V1;
        V0.q[0] = *(const uint4*)(vp0);  V0.q[1] = *(const uint4*)(vp0 + 8);
        V1.q[0] = *(const uint4*)(vp1);  V1.q[1] = *(const uint4*)(vp1 + 8);

        if (jb + 32 < SEQ) {                     // global_prefetch_b8
            __builtin_prefetch(kp0 + 32 * HID, 0, 3);
            __builtin_prefetch(vp0 + 32, 0, 3);
        }

        v8f s0 = {}, s1 = {};
        s0 = __builtin_amdgcn_wmma_f32_16x16x32_bf16(false, Q.v, false, K0.v,
                                                     (short)0, s0, false, false);
        s1 = __builtin_amdgcn_wmma_f32_16x16x32_bf16(false, Q.v, false, K1.v,
                                                     (short)0, s1, false, false);

        const bool m0 = mask[b * SEQ + j0]      != 0;
        const bool m1 = mask[b * SEQ + j0 + 16] != 0;

#pragma unroll
        for (int r = 0; r < 8; ++r) {
            float v0 = m0 ? s0[r] * scale : -100000.0f;
            float v1 = m1 ? s1[r] * scale : -100000.0f;

            float mx = fmaxf(v0, v1);
            mx = fmaxf(mx, __shfl_xor(mx, 1));
            mx = fmaxf(mx, __shfl_xor(mx, 2));
            mx = fmaxf(mx, __shfl_xor(mx, 4));
            mx = fmaxf(mx, __shfl_xor(mx, 8));

            const float mnew  = fmaxf(mrow[r], mx);
            const float alpha = __expf(mrow[r] - mnew);
            const float p0    = __expf(v0 - mnew);
            const float p1    = __expf(v1 - mnew);

            float rs = p0 + p1;
            rs += __shfl_xor(rs, 1);
            rs += __shfl_xor(rs, 2);
            rs += __shfl_xor(rs, 4);
            rs += __shfl_xor(rs, 8);

            lrow[r] = lrow[r] * alpha + rs;
            mrow[r] = mnew;
            o0[r] *= alpha;
            o1[r] *= alpha;

            const int rr = r + (lhi << 3);
            plds[wave][rr][llo]      = p0;
            plds[wave][rr][llo + 16] = p1;
        }

        asm volatile("s_wait_dscnt 0" ::: "memory");   // same-wave LDS RAW

        BF16x16 P;
        {
            const int kb = lhi << 4;
#pragma unroll
            for (int e = 0; e < 8; ++e) {
                P.h[e]     = f2bf(plds[wave][llo][kb + e]);
                P.h[e + 8] = f2bf(plds[wave][llo][kb + 8 + e]);
            }
        }

        o0 = __builtin_amdgcn_wmma_f32_16x16x32_bf16(false, P.v, false, V0.v,
                                                     (short)0, o0, false, false);
        o1 = __builtin_amdgcn_wmma_f32_16x16x32_bf16(false, P.v, false, V1.v,
                                                     (short)0, o1, false, false);
    }

#pragma unroll
    for (int r = 0; r < 8; ++r) {
        const float inv = 1.0f / lrow[r];
        const int row = b * SEQ + qb + r + (lhi << 3);
        Obf[row * HID + h * HDIM + llo]      = f2bf(o0[r] * inv);
        Obf[row * HID + h * HDIM + llo + 16] = f2bf(o1[r] * inv);
    }
}

// ---------------------------------------------------------------------------
// Kernel 3: output projection.  out = O @ wo^T + bo  (fp32 result)
// grid = (M/512, N/32), block = 256; 4M x 2N blocking like K1.
// ---------------------------------------------------------------------------
__global__ __launch_bounds__(256) void proj_kernel(
    const unsigned short* __restrict__ Obf,
    const unsigned short* __restrict__ wobf, const float* __restrict__ bo,
    float* __restrict__ out)
{
    const int lane  = threadIdx.x & 31;
    const int wave  = threadIdx.x >> 5;
    const int Mbase = blockIdx.x * 512 + wave * 64;
    const int Nbase = blockIdx.y * 32;
    const int lhi = lane >> 4;
    const int llo = lane & 15;

    v8f acc[4][2] = {};

#pragma unroll
    for (int k0 = 0; k0 < HID; k0 += 32) {
        BF16x16 B0, B1;
        {
            const unsigned short* bp0 = wobf + (Nbase + llo)      * HID + k0 + (lhi << 4);
            const unsigned short* bp1 = wobf + (Nbase + 16 + llo) * HID + k0 + (lhi << 4);
            B0.q[0] = *(const uint4*)(bp0);  B0.q[1] = *(const uint4*)(bp0 + 8);
            B1.q[0] = *(const uint4*)(bp1);  B1.q[1] = *(const uint4*)(bp1 + 8);
        }

        const int kb = k0 + (lhi << 4);
        BF16x16 A[4];
        {
            const unsigned short* ap = Obf + (Mbase + llo) * HID + kb;
#pragma unroll
            for (int t = 0; t < 4; ++t) {
                A[t].q[0] = *(const uint4*)(ap);
                A[t].q[1] = *(const uint4*)(ap + 8);
                ap += 16 * HID;
            }
        }

#pragma unroll
        for (int t = 0; t < 4; ++t) {
            acc[t][0] = __builtin_amdgcn_wmma_f32_16x16x32_bf16(
                false, A[t].v, false, B0.v, (short)0, acc[t][0], false, false);
            acc[t][1] = __builtin_amdgcn_wmma_f32_16x16x32_bf16(
                false, A[t].v, false, B1.v, (short)0, acc[t][1], false, false);
        }
    }

#pragma unroll
    for (int u = 0; u < 2; ++u) {
        const int bcol = Nbase + u * 16 + llo;
        const float bn = bo[bcol];
#pragma unroll
        for (int t = 0; t < 4; ++t) {
#pragma unroll
            for (int r = 0; r < 8; ++r) {
                const int m = Mbase + t * 16 + r + (lhi << 3);
                out[m * HID + bcol] = acc[t][u][r] + bn;
            }
        }
    }
}

// ---------------------------------------------------------------------------
extern "C" void kernel_launch(void* const* d_in, const int* in_sizes, int n_in,
                              void* d_out, int out_size, void* d_ws, size_t ws_size,
                              hipStream_t stream)
{
    const float*         x    = (const float*)d_in[0];
    const unsigned char* mask = (const unsigned char*)d_in[1];   // bool key mask
    const float* wq = (const float*)d_in[2];
    const float* bq = (const float*)d_in[3];
    const float* wk = (const float*)d_in[4];
    const float* bk = (const float*)d_in[5];
    const float* wv = (const float*)d_in[6];
    const float* bv = (const float*)d_in[7];
    const float* wo = (const float*)d_in[8];
    const float* bo = (const float*)d_in[9];
    float* out = (float*)d_out;

    const size_t M  = (size_t)BATCH * SEQ;                 // 8192
    const size_t MB = M * HID;                             // 2,097,152 elements
    const size_t WB = (size_t)HID * HID;                   // 65,536 elements

    char* ws = (char*)d_ws;
    size_t off = 0;
    unsigned short* xbf  = (unsigned short*)(ws + off); off += MB * 2;   // 4 MB
    unsigned short* Qbf  = (unsigned short*)(ws + off); off += MB * 2;   // 4 MB
    unsigned short* Kbf  = (unsigned short*)(ws + off); off += MB * 2;   // 4 MB
    unsigned short* Vt   = (unsigned short*)(ws + off); off += MB * 2;   // 4 MB
    unsigned short* Obf  = (unsigned short*)(ws + off); off += MB * 2;   // 4 MB
    unsigned short* wqbf = (unsigned short*)(ws + off); off += WB * 2;
    unsigned short* wkbf = (unsigned short*)(ws + off); off += WB * 2;
    unsigned short* wvbf = (unsigned short*)(ws + off); off += WB * 2;
    unsigned short* wobf = (unsigned short*)(ws + off); off += WB * 2;

    dim3 blk(256);

    // K0: one-time fp32 -> bf16 conversions
    cvt_kernel<<<dim3((MB / 4 + 255) / 256), blk, 0, stream>>>(x,  xbf,  (int)(MB / 4));
    cvt_kernel<<<dim3((WB / 4 + 255) / 256), blk, 0, stream>>>(wq, wqbf, (int)(WB / 4));
    cvt_kernel<<<dim3((WB / 4 + 255) / 256), blk, 0, stream>>>(wk, wkbf, (int)(WB / 4));
    cvt_kernel<<<dim3((WB / 4 + 255) / 256), blk, 0, stream>>>(wv, wvbf, (int)(WB / 4));
    cvt_kernel<<<dim3((WB / 4 + 255) / 256), blk, 0, stream>>>(wo, wobf, (int)(WB / 4));

    // K1: QKV projection (Q, K row-major bf16; V transposed bf16)
    qkv_kernel<<<dim3(M / 512, HID / 32, 3), blk, 0, stream>>>(
        xbf, wqbf, bq, wkbf, bk, wvbf, bv, Qbf, Kbf, Vt);

    // K2: flash attention per (batch, head)
    attn_kernel<<<dim3(SEQ / 128, BATCH * NHEAD), blk, 0, stream>>>(
        Qbf, Kbf, Vt, mask, Obf);

    // K3: output projection
    proj_kernel<<<dim3(M / 512, HID / 32), blk, 0, stream>>>(
        Obf, wobf, bo, out);
}
